// MultiHeadAttention_60129542144000
// MI455X (gfx1250) — matrix operand order repaired and checked by reading, not compiled, not run on hardware
//
#include <hip/hip_runtime.h>
#include <stdint.h>

#define D_MODEL 1024
#define NUM_HEADS 16
#define DK 64
#define BATCH 2
#define SEQ 2048
#define MTOT (BATCH * SEQ)   // 4096 tokens total

typedef __bf16 bf16;
typedef __attribute__((ext_vector_type(16))) __bf16 v16bf;
typedef __attribute__((ext_vector_type(8)))  __bf16 bf16x8;
typedef __attribute__((ext_vector_type(8)))  float  v8f;

// ---------------------------------------------------------------- helpers

static __device__ __forceinline__ v8f wmma_bf16(v16bf a, v16bf b, v8f c) {
  // D = A(16x32 bf16) * B(32x16 bf16) + C(16x16 f32)
  return __builtin_amdgcn_wmma_f32_16x16x32_bf16(false, a, false, b,
                                                 (short)0, c, false, false);
}

static __device__ __forceinline__ v8f vzero8() {
  v8f z;
#pragma unroll
  for (int i = 0; i < 8; ++i) z[i] = 0.0f;
  return z;
}

// A-matrix 16x32 bf16 layout (ISA 7.12.2): lane L: m = L%16;
// elements 0..7  -> K = (L/16)*8 + e        (VGPR 0..3)
// elements 8..15 -> K = 16 + (L/16)*8 + e-8 (VGPR 4..7)
static __device__ __forceinline__ v16bf packA_bf16(const bf16* row, int k0, int hi) {
  bf16x8 c0 = *reinterpret_cast<const bf16x8*>(row + k0 + hi * 8);
  bf16x8 c1 = *reinterpret_cast<const bf16x8*>(row + k0 + 16 + hi * 8);
  v16bf a;
#pragma unroll
  for (int i = 0; i < 8; ++i) { a[i] = c0[i]; a[8 + i] = c1[i]; }
  return a;
}

// B-matrix 32x16 bf16 layout: lane L: n = L%16; e<8 -> K = (L/16)*8 + e, e>=8 -> K = 16 + (L/16)*8 + e-8
static __device__ __forceinline__ v16bf packB_bf16(const bf16* row, int k0, int hi) {
  bf16x8 c0 = *reinterpret_cast<const bf16x8*>(row + k0 + hi * 8);
  bf16x8 c1 = *reinterpret_cast<const bf16x8*>(row + k0 + 16 + hi * 8);
  v16bf b;
#pragma unroll
  for (int i = 0; i < 8; ++i) { b[i] = c0[i]; b[8 + i] = c1[i]; }
  return b;
}

static __device__ __forceinline__ v16bf packA_f32(const float* row, int k0, int hi) {
  const float4* p0 = reinterpret_cast<const float4*>(row + k0 + hi * 8);
  const float4* p1 = reinterpret_cast<const float4*>(row + k0 + 16 + hi * 8);
  float4 x0 = p0[0], x1 = p0[1], x2 = p1[0], x3 = p1[1];
  v16bf a;
  a[0]=(bf16)x0.x; a[1]=(bf16)x0.y; a[2]=(bf16)x0.z; a[3]=(bf16)x0.w;
  a[4]=(bf16)x1.x; a[5]=(bf16)x1.y; a[6]=(bf16)x1.z; a[7]=(bf16)x1.w;
  a[8]=(bf16)x2.x; a[9]=(bf16)x2.y; a[10]=(bf16)x2.z; a[11]=(bf16)x2.w;
  a[12]=(bf16)x3.x; a[13]=(bf16)x3.y; a[14]=(bf16)x3.z; a[15]=(bf16)x3.w;
  return a;
}

static __device__ __forceinline__ v16bf packB_f32(const float* row, int k0, int hi) {
  const float4* p0 = reinterpret_cast<const float4*>(row + k0 + hi * 8);
  const float4* p1 = reinterpret_cast<const float4*>(row + k0 + 16 + hi * 8);
  float4 x0 = p0[0], x1 = p0[1], x2 = p1[0], x3 = p1[1];
  v16bf b;
  b[0]=(bf16)x0.x; b[1]=(bf16)x0.y; b[2]=(bf16)x0.z; b[3]=(bf16)x0.w;
  b[4]=(bf16)x1.x; b[5]=(bf16)x1.y; b[6]=(bf16)x1.z; b[7]=(bf16)x1.w;
  b[8]=(bf16)x2.x; b[9]=(bf16)x2.y; b[10]=(bf16)x2.z; b[11]=(bf16)x2.w;
  b[12]=(bf16)x3.x; b[13]=(bf16)x3.y; b[14]=(bf16)x3.z; b[15]=(bf16)x3.w;
  return b;
}

// Cross-lane reductions within a 16-lane half (xor of bits 0..3 stays in group).
static __device__ __forceinline__ float redmax16(float v) {
  v = fmaxf(v, __shfl_xor(v, 1, 32));
  v = fmaxf(v, __shfl_xor(v, 2, 32));
  v = fmaxf(v, __shfl_xor(v, 4, 32));
  v = fmaxf(v, __shfl_xor(v, 8, 32));
  return v;
}
static __device__ __forceinline__ float redsum16(float v) {
  v += __shfl_xor(v, 1, 32);
  v += __shfl_xor(v, 2, 32);
  v += __shfl_xor(v, 4, 32);
  v += __shfl_xor(v, 8, 32);
  return v;
}

// ---------------------------------------------------------------- kernels

// Fuse attention_mask [B,T,T] and key_padding_mask [B,T] into one byte tensor.
__global__ void build_mask_kernel(const int* __restrict__ am,
                                  const int* __restrict__ kpm,
                                  uint8_t* __restrict__ m8) {
  size_t idx = (size_t)blockIdx.x * blockDim.x + threadIdx.x;
  const size_t tot = (size_t)BATCH * SEQ * SEQ;
  if (idx >= tot) return;
  const int b  = (int)(idx / ((size_t)SEQ * SEQ));
  const int tk = (int)(idx % SEQ);
  m8[idx] = (uint8_t)((am[idx] == 1) || (kpm[b * SEQ + tk] == 1));
}

// Q/K/V projections: y = x @ W^T + b, 2x2 WMMA tiles (32x32 output) per wave
// for 2x A/B fragment reuse. Written as bf16 in attention-friendly layouts:
// mode 0 -> qb[B,H,T,DK], mode 1 -> kb[B,H,T,DK], mode 2 -> vtb[B,H,DK,T].
__global__ void __launch_bounds__(32)
proj_qkv_kernel(const float* __restrict__ xq, const float* __restrict__ xk,
                const float* __restrict__ xv,
                const float* __restrict__ Wq, const float* __restrict__ bq,
                const float* __restrict__ Wk, const float* __restrict__ bk,
                const float* __restrict__ Wv, const float* __restrict__ bv,
                bf16* __restrict__ qb, bf16* __restrict__ kb,
                bf16* __restrict__ vtb) {
  const int mode = blockIdx.z;
  const float* X; const float* W; const float* bias;
  if (mode == 0)      { X = xq; W = Wq; bias = bq; }
  else if (mode == 1) { X = xk; W = Wk; bias = bk; }
  else                { X = xv; W = Wv; bias = bv; }

  const int m0 = blockIdx.x * 32;
  const int n0 = blockIdx.y * 32;
  const int lane = (int)threadIdx.x;
  const int lo = lane & 15, hi = lane >> 4;

  const float* arow0 = X + (size_t)(m0 + lo) * D_MODEL;       // contiguous in k
  const float* arow1 = X + (size_t)(m0 + 16 + lo) * D_MODEL;
  const float* brow0 = W + (size_t)(n0 + lo) * D_MODEL;       // B[k,n]=W[n,k]
  const float* brow1 = W + (size_t)(n0 + 16 + lo) * D_MODEL;

  v8f acc00 = vzero8(), acc01 = vzero8(), acc10 = vzero8(), acc11 = vzero8();
#pragma unroll 2
  for (int k0 = 0; k0 < D_MODEL; k0 += 32) {
    const v16bf a0 = packA_f32(arow0, k0, hi);
    const v16bf a1 = packA_f32(arow1, k0, hi);
    const v16bf b0 = packB_f32(brow0, k0, hi);
    const v16bf b1 = packB_f32(brow1, k0, hi);
    acc00 = wmma_bf16(a0, b0, acc00);
    acc01 = wmma_bf16(a0, b1, acc01);
    acc10 = wmma_bf16(a1, b0, acc10);
    acc11 = wmma_bf16(a1, b1, acc11);
  }

  v8f accs[2][2] = {{acc00, acc01}, {acc10, acc11}};
#pragma unroll
  for (int mt = 0; mt < 2; ++mt) {
#pragma unroll
    for (int nt = 0; nt < 2; ++nt) {
      const v8f acc = accs[mt][nt];
      const int ncol = n0 + nt * 16 + lo;
      const int h = ncol >> 6, d = ncol & 63;
      const float bias_n = bias[ncol];
#pragma unroll
      for (int r = 0; r < 8; ++r) {
        const int mg = m0 + mt * 16 + r + hi * 8;  // global token
        const int bz = mg >> 11;                   // / SEQ
        const int t  = mg & (SEQ - 1);
        const float val = acc[r] + bias_n;
        if (mode == 2) {
          vtb[(((size_t)bz * NUM_HEADS + h) * DK + d) * SEQ + t] = (bf16)val;
        } else {
          bf16* dst = (mode == 0) ? qb : kb;
          dst[(((size_t)bz * NUM_HEADS + h) * SEQ + t) * DK + d] = (bf16)val;
        }
      }
    }
  }
}

// Flash attention: one wave per (b, h, 16-query tile); 32 keys per iteration.
__global__ void __launch_bounds__(32)
flash_attn_kernel(const bf16* __restrict__ qb, const bf16* __restrict__ kb,
                  const bf16* __restrict__ vtb, const uint8_t* __restrict__ m8,
                  bf16* __restrict__ ctx) {
  const int tq0 = blockIdx.x * 16;
  const int h = blockIdx.y;
  const int b = blockIdx.z;
  const size_t bh = (size_t)b * NUM_HEADS + h;

  const bf16* qbase = qb + bh * SEQ * DK;
  const bf16* kbase = kb + bh * SEQ * DK;
  const bf16* vbase = vtb + bh * DK * SEQ;     // [DK][SEQ]
  const uint8_t* mbase = m8 + (size_t)b * SEQ * SEQ;

  const int lane = (int)threadIdx.x;
  const int lo = lane & 15, hi = lane >> 4;

  // Q tile 16x64 held as two A-matrices (d 0..31, 32..63)
  const bf16* qrow = qbase + (size_t)(tq0 + lo) * DK;
  const v16bf qa0 = packA_bf16(qrow, 0, hi);
  const v16bf qa1 = packA_bf16(qrow, 32, hi);

  __shared__ __align__(16) bf16 pls[16][40];   // P tile staging (C-layout -> A-layout)

  v8f o0 = vzero8(), o1 = vzero8(), o2 = vzero8(), o3 = vzero8();
  float mrun[8], lrun[8];
#pragma unroll
  for (int r = 0; r < 8; ++r) { mrun[r] = -3.0e38f; lrun[r] = 0.0f; }

  const float scale = 0.125f;                  // 1/sqrt(DK)
  const float NEGBIG = -1.0e30f;

  for (int n0 = 0; n0 < SEQ; n0 += 32) {
    // Prefetch next iteration's K rows (emits global_prefetch_b8; speculative,
    // address stays inside the workspace allocation).
    if (n0 + 32 < SEQ) {
      __builtin_prefetch(kbase + (size_t)(n0 + 32 + lo) * DK, 0, 0);
      __builtin_prefetch(kbase + (size_t)(n0 + 48 + lo) * DK, 0, 0);
      __builtin_prefetch(vbase + (size_t)(lo * 4) * SEQ + n0 + 32, 0, 0);
    }

    // --- scores S = Q K^T for 32 keys (two 16-key subtiles) ---
    v8f s0 = vzero8(), s1 = vzero8();
    {
      const bf16* kr = kbase + (size_t)(n0 + lo) * DK;       // B[d,n]=K[n,d]
      s0 = wmma_bf16(qa0, packB_bf16(kr, 0, hi), s0);
      s0 = wmma_bf16(qa1, packB_bf16(kr, 32, hi), s0);
    }
    {
      const bf16* kr = kbase + (size_t)(n0 + 16 + lo) * DK;
      s1 = wmma_bf16(qa0, packB_bf16(kr, 0, hi), s1);
      s1 = wmma_bf16(qa1, packB_bf16(kr, 32, hi), s1);
    }

    // --- mask + online softmax (row r+hi*8 lives across the 16-lane half) ---
#pragma unroll
    for (int r = 0; r < 8; ++r) {
      const int row = tq0 + r + hi * 8;
      const uint8_t* mr = mbase + (size_t)row * SEQ + n0;
      const float a = mr[lo]      ? NEGBIG : s0[r] * scale;
      const float c = mr[16 + lo] ? NEGBIG : s1[r] * scale;
      const float mx = redmax16(fmaxf(a, c));
      const float mnew = fmaxf(mrun[r], mx);
      const float alpha = __expf(mrun[r] - mnew);
      mrun[r] = mnew;
      const float p0 = __expf(a - mnew);
      const float p1 = __expf(c - mnew);
      pls[r + hi * 8][lo]      = (bf16)p0;
      pls[r + hi * 8][16 + lo] = (bf16)p1;
      lrun[r] = lrun[r] * alpha + redsum16(p0 + p1);
      o0[r] *= alpha; o1[r] *= alpha; o2[r] *= alpha; o3[r] *= alpha;
    }

    // --- O += P V  (P from LDS as A-matrix; V transposed -> contiguous B loads) ---
    // Same-wave LDS ops are in-order; no barrier needed with one wave.
    const v16bf pa = packA_bf16(&pls[lo][0], 0, hi);
    o0 = wmma_bf16(pa, packB_bf16(vbase + (size_t)(0 * 16 + lo) * SEQ, n0, hi), o0);
    o1 = wmma_bf16(pa, packB_bf16(vbase + (size_t)(1 * 16 + lo) * SEQ, n0, hi), o1);
    o2 = wmma_bf16(pa, packB_bf16(vbase + (size_t)(2 * 16 + lo) * SEQ, n0, hi), o2);
    o3 = wmma_bf16(pa, packB_bf16(vbase + (size_t)(3 * 16 + lo) * SEQ, n0, hi), o3);
  }

  // --- normalize and store context as [B,T,D_MODEL] bf16 (A-layout friendly) ---
#pragma unroll
  for (int r = 0; r < 8; ++r) {
    const float inv = 1.0f / lrun[r];
    const int row = tq0 + r + hi * 8;
    bf16* orow = ctx + ((size_t)b * SEQ + row) * D_MODEL + h * DK;
    orow[0 * 16 + lo] = (bf16)(o0[r] * inv);
    orow[1 * 16 + lo] = (bf16)(o1[r] * inv);
    orow[2 * 16 + lo] = (bf16)(o2[r] * inv);
    orow[3 * 16 + lo] = (bf16)(o3[r] * inv);
  }
}

// Output projection: out = ctx @ Wo^T + bo (f32 result), 2x2 tiles per wave.
__global__ void __launch_bounds__(32)
proj_out_kernel(const bf16* __restrict__ ctx, const float* __restrict__ Wo,
                const float* __restrict__ bo, float* __restrict__ out) {
  const int m0 = blockIdx.x * 32;
  const int n0 = blockIdx.y * 32;
  const int lane = (int)threadIdx.x;
  const int lo = lane & 15, hi = lane >> 4;

  const bf16* arow0 = ctx + (size_t)(m0 + lo) * D_MODEL;
  const bf16* arow1 = ctx + (size_t)(m0 + 16 + lo) * D_MODEL;
  const float* brow0 = Wo + (size_t)(n0 + lo) * D_MODEL;
  const float* brow1 = Wo + (size_t)(n0 + 16 + lo) * D_MODEL;

  v8f acc00 = vzero8(), acc01 = vzero8(), acc10 = vzero8(), acc11 = vzero8();
#pragma unroll 2
  for (int k0 = 0; k0 < D_MODEL; k0 += 32) {
    const v16bf a0 = packA_bf16(arow0, k0, hi);
    const v16bf a1 = packA_bf16(arow1, k0, hi);
    const v16bf b0 = packB_f32(brow0, k0, hi);
    const v16bf b1 = packB_f32(brow1, k0, hi);
    acc00 = wmma_bf16(a0, b0, acc00);
    acc01 = wmma_bf16(a0, b1, acc01);
    acc10 = wmma_bf16(a1, b0, acc10);
    acc11 = wmma_bf16(a1, b1, acc11);
  }

  v8f accs[2][2] = {{acc00, acc01}, {acc10, acc11}};
#pragma unroll
  for (int mt = 0; mt < 2; ++mt) {
#pragma unroll
    for (int nt = 0; nt < 2; ++nt) {
      const v8f acc = accs[mt][nt];
      const int ncol = n0 + nt * 16 + lo;
      const float bias_n = bo[ncol];
#pragma unroll
      for (int r = 0; r < 8; ++r) {
        const int mg = m0 + mt * 16 + r + hi * 8;
        out[(size_t)mg * D_MODEL + ncol] = acc[r] + bias_n;
      }
    }
  }
}

// ---------------------------------------------------------------- launch

extern "C" void kernel_launch(void* const* d_in, const int* in_sizes, int n_in,
                              void* d_out, int out_size, void* d_ws, size_t ws_size,
                              hipStream_t stream) {
  (void)in_sizes; (void)n_in; (void)out_size; (void)ws_size;

  const float* q_in = (const float*)d_in[0];
  const float* k_in = (const float*)d_in[1];
  const float* v_in = (const float*)d_in[2];
  const int*   kpm  = (const int*)d_in[3];
  const int*   am   = (const int*)d_in[4];
  const float* Wq = (const float*)d_in[5];  const float* bq = (const float*)d_in[6];
  const float* Wk = (const float*)d_in[7];  const float* bk = (const float*)d_in[8];
  const float* Wv = (const float*)d_in[9];  const float* bv = (const float*)d_in[10];
  const float* Wo = (const float*)d_in[11]; const float* bo = (const float*)d_in[12];
  float* out = (float*)d_out;

  // Workspace carve-up (bf16 tensors, then byte mask): ~40 MB total.
  const size_t nelem = (size_t)MTOT * D_MODEL;   // 4,194,304
  bf16* qb  = (bf16*)d_ws;
  bf16* kb  = qb + nelem;
  bf16* vtb = kb + nelem;
  bf16* ctx = vtb + nelem;
  uint8_t* m8 = (uint8_t*)(ctx + nelem);

  {
    const size_t tot = (size_t)BATCH * SEQ * SEQ;
    const int blocks = (int)((tot + 255) / 256);
    build_mask_kernel<<<blocks, 256, 0, stream>>>(am, kpm, m8);
  }
  proj_qkv_kernel<<<dim3(MTOT / 32, D_MODEL / 32, 3), 32, 0, stream>>>(
      q_in, k_in, v_in, Wq, bq, Wk, bk, Wv, bv, qb, kb, vtb);
  flash_attn_kernel<<<dim3(SEQ / 16, NUM_HEADS, BATCH), 32, 0, stream>>>(
      qb, kb, vtb, m8, ctx);
  proj_out_kernel<<<dim3(MTOT / 32, D_MODEL / 32), 32, 0, stream>>>(
      ctx, Wo, bo, out);
}
